// GINDirLayer_65532611002908
// MI455X (gfx1250) — matrix operand order repaired and checked by reading, not compiled, not run on hardware
//
#include <hip/hip_runtime.h>
#include <math.h>

typedef __attribute__((ext_vector_type(16))) __bf16 v16bf;
typedef __attribute__((ext_vector_type(8)))  float  v8f;
typedef __attribute__((ext_vector_type(4)))  float  v4f;

#define IN_DIM   64
#define EMBED    128
#define EDGE_DIM 32

// ---- CDNA5 WMMA fragment index helpers (wave32, 16x16x32 bf16) ----
// A-matrix (16x32, 16-bit): lane L holds row M = L&15; half = L>>4.
// VGPR j holds K pair: K = 2j + (j>=4 ? 8 : 0) + 8*half (+0/+1).
__device__ __forceinline__ int a_k(int i, int half) {
  int j = i >> 1, p = i & 1;
  return 2 * j + ((j >= 4) ? 8 : 0) + 8 * half + p;
}

__device__ __forceinline__ v8f wmma_bf16(v16bf a, v16bf b, v8f c) {
  return __builtin_amdgcn_wmma_f32_16x16x32_bf16(false, a, false, b, (short)0, c,
                                                 false, false);
}

__global__ __launch_bounds__(256)
void zero_f32(v4f* __restrict__ p, int n4) {
  int i = blockIdx.x * blockDim.x + threadIdx.x;
  int stride = gridDim.x * blockDim.x;
  v4f z = {0.f, 0.f, 0.f, 0.f};
  for (; i < n4; i += stride) p[i] = z;
}

// ---------------------------------------------------------------------------
// Fused edge stage, BOTH directions. One wave = 16 edges.
// edge_attr / edge_index are streamed NON-TEMPORAL (read exactly once; keeps
// x/agg/out resident in the 192MB L2). Gathers are batched so the compiler
// can issue load clauses; scatter via global_atomic_add_f32 (L2-resident).
// ---------------------------------------------------------------------------
__global__ __launch_bounds__(256)
void gine_edge_kernel(const float* __restrict__ x,
                      const int* __restrict__ ei32,        // int64 viewed as int pairs
                      const float* __restrict__ ea,        // [E][32]
                      const float* __restrict__ We_in,  const float* __restrict__ be_in,
                      const float* __restrict__ We_out, const float* __restrict__ be_out,
                      float* __restrict__ agg_in, float* __restrict__ agg_out,
                      int E)
{
  __shared__ float sA[8][16][EDGE_DIM];        // 16 KB per-wave A staging

  const int tid  = threadIdx.x;
  const int wid  = tid >> 5;
  const int lane = tid & 31;
  const int half = lane >> 4;
  const int mrow = lane & 15;

  // Hoist B fragments of both edge-linear weights (col = nt*16 + lane&15; K row = a_k(i, half)).
  v16bf bIn[4], bOut[4];
  #pragma unroll
  for (int nt = 0; nt < 4; ++nt) {
    const float* wi = We_in  + nt * 16 + mrow;
    const float* wo = We_out + nt * 16 + mrow;
    #pragma unroll
    for (int i = 0; i < 16; ++i) {
      bIn[nt][i]  = (__bf16)wi[a_k(i, half) * IN_DIM];
      bOut[nt][i] = (__bf16)wo[a_k(i, half) * IN_DIM];
    }
  }
  // Hoist per-lane bias values (col = nt*16 + mrow).
  float beI[4], beO[4];
  #pragma unroll
  for (int nt = 0; nt < 4; ++nt) {
    beI[nt] = be_in[nt * 16 + mrow];
    beO[nt] = be_out[nt * 16 + mrow];
  }

  const int tiles = E >> 4;
  for (int t = blockIdx.x * 8 + wid; t < tiles; t += gridDim.x * 8) {
    const int e0 = t << 4;

    // stage 16x32 f32 tile of edge_attr (coalesced v4f, non-temporal)
    #pragma unroll
    for (int q = 0; q < 4; ++q) {
      int f4  = lane + q * 32;
      int row = f4 >> 3, c4 = (f4 & 7) << 2;
      v4f v = __builtin_nontemporal_load(
          (const v4f*)(ea + (size_t)(e0 + row) * EDGE_DIM + c4));
      *(v4f*)&sA[wid][row][c4] = v;
    }
    __builtin_amdgcn_wave_barrier();   // LDS in-order within a wave

    // A fragment (16x32 bf16), shared by both directions
    v16bf a;
    #pragma unroll
    for (int i = 0; i < 16; ++i) a[i] = (__bf16)sA[wid][mrow][a_k(i, half)];

    // Edge endpoints: low dword of int64 (fits i32); rows M = r + 8*half.
    int s32[8], d32[8];
    #pragma unroll
    for (int r = 0; r < 8; ++r) {
      int e = e0 + r + 8 * half;
      s32[r] = __builtin_nontemporal_load(&ei32[2 * e]);
      d32[r] = __builtin_nontemporal_load(&ei32[2 * (E + e)]);
    }

    // ---- direction "in": relu(x[src] + ea@We_in + be_in) -> agg_in[dst]
    {
      v8f c[4];
      #pragma unroll
      for (int nt = 0; nt < 4; ++nt) {
        v8f z = {0.f, 0.f, 0.f, 0.f, 0.f, 0.f, 0.f, 0.f};
        c[nt] = wmma_bf16(a, bIn[nt], z);
      }
      // batch all 32 gathers first (one load clause, latency overlapped)
      float xv[4][8];
      #pragma unroll
      for (int nt = 0; nt < 4; ++nt) {
        int col = nt * 16 + mrow;
        #pragma unroll
        for (int r = 0; r < 8; ++r)
          xv[nt][r] = x[(unsigned)s32[r] * IN_DIM + col];
      }
      #pragma unroll
      for (int nt = 0; nt < 4; ++nt) {
        int col = nt * 16 + mrow;
        #pragma unroll
        for (int r = 0; r < 8; ++r) {
          float v = c[nt][r] + beI[nt] + xv[nt][r];
          v = v > 0.f ? v : 0.f;
          atomicAdd(agg_in + (unsigned)d32[r] * IN_DIM + col, v);
        }
      }
    }
    // ---- direction "out": relu(x[dst] + ea@We_out + be_out) -> agg_out[src]
    {
      v8f c[4];
      #pragma unroll
      for (int nt = 0; nt < 4; ++nt) {
        v8f z = {0.f, 0.f, 0.f, 0.f, 0.f, 0.f, 0.f, 0.f};
        c[nt] = wmma_bf16(a, bOut[nt], z);
      }
      float xv[4][8];
      #pragma unroll
      for (int nt = 0; nt < 4; ++nt) {
        int col = nt * 16 + mrow;
        #pragma unroll
        for (int r = 0; r < 8; ++r)
          xv[nt][r] = x[(unsigned)d32[r] * IN_DIM + col];
      }
      #pragma unroll
      for (int nt = 0; nt < 4; ++nt) {
        int col = nt * 16 + mrow;
        #pragma unroll
        for (int r = 0; r < 8; ++r) {
          float v = c[nt][r] + beO[nt] + xv[nt][r];
          v = v > 0.f ? v : 0.f;
          atomicAdd(agg_out + (unsigned)s32[r] * IN_DIM + col, v);
        }
      }
    }
  }
}

// ---------------------------------------------------------------------------
// Root: out = x @ Wr + br. Wr fragments live entirely in registers (128 VGPRs).
// ---------------------------------------------------------------------------
__global__ __launch_bounds__(256)
void gine_root_kernel(const float* __restrict__ x,
                      const float* __restrict__ Wr, const float* __restrict__ br,
                      float* __restrict__ out, int Nn)
{
  const int tid  = threadIdx.x;
  const int wid  = tid >> 5, lane = tid & 31;
  const int half = lane >> 4, mrow = lane & 15;

  v16bf wrf[2][8];                       // B frags: K row = 32*ks + a_k(i, half), col = nt*16 + mrow
  #pragma unroll
  for (int ks = 0; ks < 2; ++ks)
    #pragma unroll
    for (int nt = 0; nt < 8; ++nt) {
      const float* w = Wr + (size_t)(32 * ks) * EMBED + nt * 16 + mrow;
      #pragma unroll
      for (int i = 0; i < 16; ++i) wrf[ks][nt][i] = (__bf16)w[a_k(i, half) * EMBED];
    }
  float bb[8];
  #pragma unroll
  for (int nt = 0; nt < 8; ++nt) bb[nt] = br[nt * 16 + mrow];

  const int tiles = Nn >> 4;
  for (int t = blockIdx.x * 8 + wid; t < tiles; t += gridDim.x * 8) {
    const int n0 = t << 4;
    unsigned rowoff = (unsigned)(n0 + mrow) * IN_DIM;

    v16bf a0, a1;
    #pragma unroll
    for (int i = 0; i < 16; ++i) {
      int k = a_k(i, half);
      a0[i] = (__bf16)x[rowoff + k];
      a1[i] = (__bf16)x[rowoff + k + 32];
    }
    #pragma unroll
    for (int nt = 0; nt < 8; ++nt) {
      v8f acc = {0.f, 0.f, 0.f, 0.f, 0.f, 0.f, 0.f, 0.f};
      acc = wmma_bf16(a0, wrf[0][nt], acc);
      acc = wmma_bf16(a1, wrf[1][nt], acc);
      int col = nt * 16 + mrow;
      #pragma unroll
      for (int r = 0; r < 8; ++r)
        out[(unsigned)(n0 + r + 8 * half) * EMBED + col] = acc[r] + bb[nt];
    }
  }
}

// ---------------------------------------------------------------------------
// Node MLP, one direction: out += scale * (gelu((x+agg)@W1+b1) @ W2 + b2)
// 128 threads = 4 waves; W1(16K) + W2(32K) + per-wave transpose(16K) = 64 KB LDS.
// ---------------------------------------------------------------------------
__global__ __launch_bounds__(128)
void gine_node_kernel(const float* __restrict__ x, const float* __restrict__ agg,
                      const float* __restrict__ W1, const float* __restrict__ b1,
                      const float* __restrict__ W2, const float* __restrict__ b2,
                      float* __restrict__ out, float scale, int Nn)
{
  __shared__ __bf16 sW1[IN_DIM][EMBED];   // 16 KB
  __shared__ __bf16 sW2[EMBED][EMBED];    // 32 KB
  __shared__ __bf16 sT[4][16][EMBED];     // 16 KB (per-wave gelu output)

  for (int idx = threadIdx.x; idx < IN_DIM * EMBED; idx += 128)
    sW1[idx / EMBED][idx % EMBED] = (__bf16)W1[(((idx / EMBED) & ~31) + a_k(idx & 15, ((idx / EMBED) >> 4) & 1)) * EMBED + ((idx % EMBED) & ~15) + ((idx / EMBED) & 15)];
  for (int idx = threadIdx.x; idx < EMBED * EMBED; idx += 128)
    sW2[idx / EMBED][idx % EMBED] = (__bf16)W2[(((idx / EMBED) & ~31) + a_k(idx & 15, ((idx / EMBED) >> 4) & 1)) * EMBED + ((idx % EMBED) & ~15) + ((idx / EMBED) & 15)];
  __syncthreads();

  const int wid  = threadIdx.x >> 5, lane = threadIdx.x & 31;
  const int half = lane >> 4,        mrow = lane & 15;

  float bb1[8], bb2[8];
  #pragma unroll
  for (int nt = 0; nt < 8; ++nt) {
    bb1[nt] = b1[nt * 16 + mrow];
    bb2[nt] = b2[nt * 16 + mrow];
  }

  const int tiles = Nn >> 4;
  for (int t = blockIdx.x * 4 + wid; t < tiles; t += gridDim.x * 4) {
    const int n0 = t << 4;
    unsigned rowoff = (unsigned)(n0 + mrow) * IN_DIM;

    // A fragments of h = x + agg (K = 64 -> 2 chunks); all L2-resident reads
    v16bf a0, a1;
    #pragma unroll
    for (int i = 0; i < 16; ++i) {
      int k = a_k(i, half);
      a0[i] = (__bf16)(x[rowoff + k]      + agg[rowoff + k]);
      a1[i] = (__bf16)(x[rowoff + k + 32] + agg[rowoff + k + 32]);
    }

    // GEMM1 + bias + exact gelu -> sT (row-major 16x128 bf16)
    #pragma unroll
    for (int nt = 0; nt < 8; ++nt) {
      v8f acc = {0.f, 0.f, 0.f, 0.f, 0.f, 0.f, 0.f, 0.f};
      acc = wmma_bf16(a0, *(const v16bf*)&sW1[lane][nt * 16], acc);
      acc = wmma_bf16(a1, *(const v16bf*)&sW1[32 + lane][nt * 16], acc);
      int col = nt * 16 + mrow;
      #pragma unroll
      for (int r = 0; r < 8; ++r) {
        float v = acc[r] + bb1[nt];
        float g = 0.5f * v * (1.0f + erff(v * 0.70710678f));
        sT[wid][r + 8 * half][col] = (__bf16)g;
      }
    }
    __builtin_amdgcn_wave_barrier();

    // A fragments of gelu output (K = 128 -> 4 chunks) from LDS
    v16bf ta[4];
    #pragma unroll
    for (int ks = 0; ks < 4; ++ks) {
      #pragma unroll
      for (int i = 0; i < 16; ++i)
        ta[ks][i] = sT[wid][mrow][32 * ks + a_k(i, half)];
    }

    // GEMM2 + bias, accumulate into out
    #pragma unroll
    for (int nt = 0; nt < 8; ++nt) {
      v8f acc = {0.f, 0.f, 0.f, 0.f, 0.f, 0.f, 0.f, 0.f};
      #pragma unroll
      for (int ks = 0; ks < 4; ++ks)
        acc = wmma_bf16(ta[ks], *(const v16bf*)&sW2[32 * ks + lane][nt * 16], acc);
      int col = nt * 16 + mrow;
      #pragma unroll
      for (int r = 0; r < 8; ++r) {
        unsigned o = (unsigned)(n0 + r + 8 * half) * EMBED + col;
        out[o] += scale * (acc[r] + bb2[nt]);
      }
    }
  }
}

// ---------------------------------------------------------------------------
extern "C" void kernel_launch(void* const* d_in, const int* in_sizes, int n_in,
                              void* d_out, int out_size, void* d_ws, size_t ws_size,
                              hipStream_t stream)
{
  const float* x      = (const float*)d_in[0];
  const int*   ei32   = (const int*)d_in[1];      // int64 pairs, low dwords used
  const float* ea     = (const float*)d_in[2];
  const float* We_in  = (const float*)d_in[3];
  const float* be_in  = (const float*)d_in[4];
  const float* W1_in  = (const float*)d_in[5];
  const float* b1_in  = (const float*)d_in[6];
  const float* W2_in  = (const float*)d_in[7];
  const float* b2_in  = (const float*)d_in[8];
  const float* We_out = (const float*)d_in[9];
  const float* be_out = (const float*)d_in[10];
  const float* W1_out = (const float*)d_in[11];
  const float* b1_out = (const float*)d_in[12];
  const float* W2_out = (const float*)d_in[13];
  const float* b2_out = (const float*)d_in[14];
  const float* Wr     = (const float*)d_in[15];
  const float* br     = (const float*)d_in[16];
  float*       out    = (float*)d_out;

  const int Nn = in_sizes[0] / IN_DIM;   // 100000
  const int E  = in_sizes[1] / 2;        // 1600000

  float* agg_in  = (float*)d_ws;                   // [N,64] f32
  float* agg_out = agg_in + (size_t)Nn * IN_DIM;   // [N,64] f32

  // 1) zero accumulators (ws is poisoned; must be deterministic every call)
  zero_f32<<<2048, 256, 0, stream>>>((v4f*)agg_in, (2 * Nn * IN_DIM) / 4);

  // 2) fused edge stage, both directions
  gine_edge_kernel<<<2048, 256, 0, stream>>>(x, ei32, ea, We_in, be_in,
                                             We_out, be_out, agg_in, agg_out, E);

  // 3) root writes out = x@Wr + br
  gine_root_kernel<<<784, 256, 0, stream>>>(x, Wr, br, out, Nn);

  // 4) out += (1-ALPHA)*h_in ; out += ALPHA*h_out   (ALPHA = 0.5)
  gine_node_kernel<<<1568, 128, 0, stream>>>(x, agg_in,  W1_in,  b1_in,
                                             W2_in,  b2_in,  out, 0.5f, Nn);
  gine_node_kernel<<<1568, 128, 0, stream>>>(x, agg_out, W1_out, b1_out,
                                             W2_out, b2_out, out, 0.5f, Nn);
}
